// GATClassifier_38809324486858
// MI455X (gfx1250) — matrix operand order repaired and checked by reading, not compiled, not run on hardware
//
#include <hip/hip_runtime.h>

#define NN 8192
#define DD 64
#define EE 524288

typedef __attribute__((ext_vector_type(16))) __bf16 v16bf;
typedef __attribute__((ext_vector_type(8)))  float  v8f;

// ---------- helpers ----------
// order-preserving float<->uint encoding for atomicMax-based segment max
__device__ __forceinline__ unsigned fenc(float f) {
  unsigned u = __float_as_uint(f);
  return (u & 0x80000000u) ? ~u : (u | 0x80000000u);
}
__device__ __forceinline__ float fdec(unsigned u) {
  unsigned b = (u & 0x80000000u) ? (u & 0x7FFFFFFFu) : ~u;
  return __uint_as_float(b);
}

// ---------- bf16 layout prep ----------
// X[K][64] (f32, row major) -> Bt[64][K] (bf16): WMMA B-fragment friendly
__global__ void transpose_to_bf16(const float* __restrict__ X,
                                  __bf16* __restrict__ Bt, int K) {
  long i = (long)blockIdx.x * blockDim.x + threadIdx.x;   // index into Bt
  long total = (long)K * DD;
  if (i >= total) return;
  int  c = (int)(i / K);
  long k = i - (long)c * K;
  Bt[i] = (__bf16)X[k * DD + c];
}

__global__ void cvt_to_bf16(const float* __restrict__ X,
                            __bf16* __restrict__ Y, int n) {
  int i = blockIdx.x * blockDim.x + threadIdx.x;
  if (i < n) Y[i] = (__bf16)X[i];
}

// ---------- WMMA GEMM: C[8192][64] (+)= A[8192][K] * B, Bt[64][K] bf16 ----------
// 4 waves/block, wave owns a 16-row tile and all 64 columns (4 WMMA col tiles).
// Split-K over gridDim.y; ATOMIC => f32 atomic output (C pre-zeroed).
// NT => nontemporal A-stream loads + prefetch (one-pass 256MB adjacency).
//
// ISA fragment layouts (CDNA5 ISA):
//  A 16x32 bf16 : lanes 0-15 -> M=lane,  K = {0..7} u {16..23}
//                 lanes 16-31 -> M=lane-16, K = {8..15} u {24..31}
//  B 32x16 bf16 : N = lane&15; this code loads A and B alike with K = 16*(lane>=16) + s
//  C/D 16x16 f32: VGPR r -> M = r + 8*(lane>=16), N = lane&15
template <bool NT, bool ATOMIC>
__global__ __launch_bounds__(128) void gemm_bf16_wmma(
    const float*  __restrict__ A,  long lda, int K,
    const __bf16* __restrict__ Bt, long ldb,
    const float*  __restrict__ bias,
    float*        __restrict__ C) {
  const int lane = threadIdx.x & 31;
  const int wave = threadIdx.x >> 5;
  const int m0   = (blockIdx.x * 4 + wave) * 16;
  const int half = lane >> 4;        // lane group 0/1
  const int lrow = lane & 15;        // A row-in-tile / B column-in-tile

  const int kchunk = K / gridDim.y;
  const int kbeg   = blockIdx.y * kchunk;
  const int kend   = kbeg + kchunk;

  v8f acc0 = {}, acc1 = {}, acc2 = {}, acc3 = {};

  // A: lane reads 16 contiguous floats at k0+half*16
  const float*  aRow  = A  + (long)(m0 + lrow) * lda + half * 16;
  // B: lane reads 16 contiguous bf16 at k0+half*16, row = coltile*16 + lrow
  const __bf16* bBase = Bt + (long)lrow * ldb + half * 16;

#pragma unroll 2
  for (int k0 = kbeg; k0 < kend; k0 += 32) {
    const float* ap = aRow + k0;
    if (NT) __builtin_prefetch(ap + 256, 0, 0);   // ~8 k-steps ahead
    v16bf af;
#pragma unroll
    for (int i = 0; i < 8; ++i) {
      float lo = NT ? __builtin_nontemporal_load(ap + i)      : ap[i];
      float hi = NT ? __builtin_nontemporal_load(ap + 8 + i) : ap[8 + i];
      af[i]     = (__bf16)lo;
      af[8 + i] = (__bf16)hi;
    }

    const __bf16* bp = bBase + k0;
    v16bf b0, b1, b2, b3;
#pragma unroll
    for (int i = 0; i < 16; ++i) b0[i] = bp[i];
#pragma unroll
    for (int i = 0; i < 16; ++i) b1[i] = bp[(long)16 * ldb + i];
#pragma unroll
    for (int i = 0; i < 16; ++i) b2[i] = bp[(long)32 * ldb + i];
#pragma unroll
    for (int i = 0; i < 16; ++i) b3[i] = bp[(long)48 * ldb + i];

    acc0 = __builtin_amdgcn_wmma_f32_16x16x32_bf16(false, af, false, b0, (short)0, acc0, false, false);
    acc1 = __builtin_amdgcn_wmma_f32_16x16x32_bf16(false, af, false, b1, (short)0, acc1, false, false);
    acc2 = __builtin_amdgcn_wmma_f32_16x16x32_bf16(false, af, false, b2, (short)0, acc2, false, false);
    acc3 = __builtin_amdgcn_wmma_f32_16x16x32_bf16(false, af, false, b3, (short)0, acc3, false, false);
  }

  const int n     = lrow;
  const int mbase = m0 + half * 8;
#pragma unroll
  for (int r = 0; r < 8; ++r) {
    long ro = (long)(mbase + r) * DD;
    if (ATOMIC) {
      atomicAdd(&C[ro +  0 + n], acc0[r]);
      atomicAdd(&C[ro + 16 + n], acc1[r]);
      atomicAdd(&C[ro + 32 + n], acc2[r]);
      atomicAdd(&C[ro + 48 + n], acc3[r]);
    } else {
      C[ro +  0 + n] = acc0[r] + (bias ? bias[ 0 + n] : 0.f);
      C[ro + 16 + n] = acc1[r] + (bias ? bias[16 + n] : 0.f);
      C[ro + 32 + n] = acc2[r] + (bias ? bias[32 + n] : 0.f);
      C[ro + 48 + n] = acc3[r] + (bias ? bias[48 + n] : 0.f);
    }
  }
}

// ---------- edge attention phase ----------
// node mask from kept edges (motifs 1,2); plain stores of 1.0 are race-safe
__global__ void nmask_kernel(const int* __restrict__ src, const int* __restrict__ dst,
                             const float* __restrict__ Amask, float* __restrict__ nmask) {
  long e = (long)blockIdx.x * blockDim.x + threadIdx.x;
  if (e >= EE) return;
  int u = src[e], v = dst[e];
  if (Amask[(long)u * NN + v] > 0.f) { nmask[u] = 1.f; nmask[v] = 1.f; }
}

// wave-per-edge: e = <ft[src], ft[dst]> / 8 ; segment max over dst
__global__ void edge_logits(const float* __restrict__ ft,
                            const int* __restrict__ src, const int* __restrict__ dst,
                            const float* __restrict__ Amask,
                            unsigned* __restrict__ menc, float* __restrict__ eval) {
  long e = (long)blockIdx.x * 8 + (threadIdx.x >> 5);
  int lane = threadIdx.x & 31;
  if (e >= EE) return;
  int u = src[e], v = dst[e];
  const float2* fu = (const float2*)(ft + (long)u * DD);
  const float2* fv = (const float2*)(ft + (long)v * DD);
  float2 a = fu[lane], b = fv[lane];
  float part = a.x * b.x + a.y * b.y;
#pragma unroll
  for (int off = 16; off; off >>= 1) part += __shfl_xor(part, off, 32);
  if (lane == 0) {
    float ev = part * 0.125f;
    eval[e] = ev;
    bool keep = Amask ? (Amask[(long)u * NN + v] > 0.f) : true;
    if (keep) atomicMax(&menc[v], fenc(ev));
  }
}

// ex = exp(e - m[dst]) for kept edges ; segment sum of ex over dst
__global__ void edge_exp(const int* __restrict__ src, const int* __restrict__ dst,
                         const float* __restrict__ Amask,
                         const unsigned* __restrict__ menc,
                         float* __restrict__ eval_to_ex, float* __restrict__ den) {
  long e = (long)blockIdx.x * blockDim.x + threadIdx.x;
  if (e >= EE) return;
  int u = src[e], v = dst[e];
  bool keep = Amask ? (Amask[(long)u * NN + v] > 0.f) : true;
  float ex = 0.f;
  if (keep) ex = expf(eval_to_ex[e] - fdec(menc[v]));
  eval_to_ex[e] = ex;
  if (ex != 0.f) atomicAdd(&den[v], ex);
}

// agg[dst] += (ex/den[dst]) * ft[src] ; wave-per-edge
__global__ void edge_aggregate(const float* __restrict__ ft,
                               const int* __restrict__ src, const int* __restrict__ dst,
                               const float* __restrict__ exb, const float* __restrict__ den,
                               float* __restrict__ agg) {
  long e = (long)blockIdx.x * 8 + (threadIdx.x >> 5);
  int lane = threadIdx.x & 31;
  if (e >= EE) return;
  float ex = exb[e];
  if (ex == 0.f) return;                         // uniform across the wave
  int u = src[e], v = dst[e];
  float alpha = ex / fmaxf(den[v], 1e-9f);
  float2 a = ((const float2*)(ft + (long)u * DD))[lane];
  atomicAdd(&agg[(long)v * DD + lane * 2 + 0], alpha * a.x);
  atomicAdd(&agg[(long)v * DD + lane * 2 + 1], alpha * a.y);
}

// masked mean over nodes: out[d] = sum_i agg[i][d]*w[i] / max(sum_i w[i], 1)
__global__ void pool_kernel(const float* __restrict__ agg,
                            const float* __restrict__ nmask, float* __restrict__ out) {
  __shared__ float ss[256], cc[256];
  int d = blockIdx.x, t = threadIdx.x;
  float s = 0.f, c = 0.f;
  for (int i = t; i < NN; i += 256) {
    float w = nmask ? nmask[i] : 1.f;
    s += agg[(long)i * DD + d] * w;
    c += w;
  }
  ss[t] = s; cc[t] = c; __syncthreads();
  for (int off = 128; off; off >>= 1) {
    if (t < off) { ss[t] += ss[t + off]; cc[t] += cc[t + off]; }
    __syncthreads();
  }
  if (t == 0) out[d] = ss[0] / fmaxf(cc[0], 1.f);
}

// out[c] = feat[192] . lin_w[:,c] + lin_b[c]
__global__ void final_linear(const float* __restrict__ feat,
                             const float* __restrict__ lw, const float* __restrict__ lb,
                             float* __restrict__ out) {
  __shared__ float r0[64], r1[64];
  int t = threadIdx.x;
  float s0 = 0.f, s1 = 0.f;
  for (int i = t; i < 3 * DD; i += 64) {
    float f = feat[i];
    s0 += f * lw[i * 2 + 0];
    s1 += f * lw[i * 2 + 1];
  }
  r0[t] = s0; r1[t] = s1; __syncthreads();
  for (int off = 32; off; off >>= 1) {
    if (t < off) { r0[t] += r0[t + off]; r1[t] += r1[t + off]; }
    __syncthreads();
  }
  if (t == 0) { out[0] = r0[0] + lb[0]; out[1] = r1[0] + lb[1]; }
}

// ---------- host orchestration ----------
extern "C" void kernel_launch(void* const* d_in, const int* in_sizes, int n_in,
                              void* d_out, int out_size, void* d_ws, size_t ws_size,
                              hipStream_t stream) {
  (void)in_sizes; (void)n_in; (void)out_size; (void)ws_size;
  const float* h     = (const float*)d_in[0];
  const float* A     = (const float*)d_in[1];
  const float* A1    = (const float*)d_in[2];
  const float* A2    = (const float*)d_in[3];
  const float* Ww    = (const float*)d_in[4];
  const float* Wb    = (const float*)d_in[5];
  const float* fc[3] = {(const float*)d_in[6], (const float*)d_in[7], (const float*)d_in[8]};
  const float* lin_w = (const float*)d_in[9];
  const float* lin_b = (const float*)d_in[10];
  const int*   src   = (const int*)d_in[11];
  const int*   dst   = (const int*)d_in[12];
  float* out = (float*)d_out;

  // workspace carve-up (256B aligned)
  char* base = (char*)d_ws;
  size_t off = 0;
  auto alloc = [&](size_t bytes) {
    void* p = base + off;
    off = (off + bytes + 255) & ~(size_t)255;
    return p;
  };
  __bf16*   Bt_h  = (__bf16*)  alloc((size_t)DD * NN * 2);
  __bf16*   Bt_z  = (__bf16*)  alloc((size_t)DD * NN * 2);
  float*    AH    = (float*)   alloc((size_t)NN * DD * 4);
  float*    Z     = (float*)   alloc((size_t)NN * DD * 4);
  float*    AZ    = (float*)   alloc((size_t)NN * DD * 4);
  float*    ft    = (float*)   alloc((size_t)NN * DD * 4);
  __bf16*   Wsml  = (__bf16*)  alloc((size_t)DD * DD * 2);   // Ww / fc_k^T bf16
  float*    evx   = (float*)   alloc((size_t)EE * 4);        // e-values, then ex
  unsigned* menc  = (unsigned*)alloc((size_t)NN * 4);
  float*    den   = (float*)   alloc((size_t)NN * 4);
  float*    nmask = (float*)   alloc((size_t)NN * 4);
  float*    agg   = (float*)   alloc((size_t)NN * DD * 4);
  float*    feat  = (float*)   alloc((size_t)3 * DD * 4);

  const dim3 gemmBig(NN / 64, 8);     // split-K=8 -> 1024 workgroups
  const dim3 gemmSml(NN / 64, 1);
  const int  tB = 128;

  // Z = (A @ h) @ Ww.T + Wb
  transpose_to_bf16<<<(NN * DD + 255) / 256, 256, 0, stream>>>(h, Bt_h, NN);
  hipMemsetAsync(AH, 0, (size_t)NN * DD * 4, stream);
  gemm_bf16_wmma<true, true><<<gemmBig, tB, 0, stream>>>(A, NN, NN, Bt_h, NN, nullptr, AH);
  cvt_to_bf16<<<(DD * DD + 255) / 256, 256, 0, stream>>>(Ww, Wsml, DD * DD); // Bt of Ww.T is Ww
  gemm_bf16_wmma<false, false><<<gemmSml, tB, 0, stream>>>(AH, DD, DD, Wsml, DD, Wb, Z);
  transpose_to_bf16<<<(NN * DD + 255) / 256, 256, 0, stream>>>(Z, Bt_z, NN);

  const float* Amot[3] = {nullptr, A1, A2};
  for (int m = 0; m < 3; ++m) {
    const float* az = Z;
    if (m > 0) {
      hipMemsetAsync(AZ, 0, (size_t)NN * DD * 4, stream);
      gemm_bf16_wmma<true, true><<<gemmBig, tB, 0, stream>>>(Amot[m], NN, NN, Bt_z, NN, nullptr, AZ);
      az = AZ;
    }
    // ft = az @ fc_m   (Bt = fc_m^T in bf16)
    transpose_to_bf16<<<(DD * DD + 255) / 256, 256, 0, stream>>>(fc[m], Wsml, DD);
    gemm_bf16_wmma<false, false><<<gemmSml, tB, 0, stream>>>(az, DD, DD, Wsml, DD, nullptr, ft);

    hipMemsetAsync(menc, 0, (size_t)NN * 4, stream);
    hipMemsetAsync(den,  0, (size_t)NN * 4, stream);
    hipMemsetAsync(agg,  0, (size_t)NN * DD * 4, stream);
    const float* nm = nullptr;
    if (m > 0) {
      hipMemsetAsync(nmask, 0, (size_t)NN * 4, stream);
      nmask_kernel<<<EE / 256, 256, 0, stream>>>(src, dst, Amot[m], nmask);
      nm = nmask;
    }
    edge_logits   <<<EE / 8, 256, 0, stream>>>(ft, src, dst, Amot[m], menc, evx);
    edge_exp      <<<EE / 256, 256, 0, stream>>>(src, dst, Amot[m], menc, evx, den);
    edge_aggregate<<<EE / 8, 256, 0, stream>>>(ft, src, dst, evx, den, agg);
    pool_kernel   <<<DD, 256, 0, stream>>>(agg, nm, feat + m * DD);
  }

  final_linear<<<1, 64, 0, stream>>>(feat, lin_w, lin_b, out);
}
